// CausalSelfAttention_28011776705073
// MI455X (gfx1250) — matrix operand order repaired and checked by reading, not compiled, not run on hardware
//
#include <hip/hip_runtime.h>
#include <hip/hip_bf16.h>

// ---------------------------------------------------------------------------
// CDNA5 (gfx1250) causal self-attention, bf16 WMMA pipeline.
// B=4, T=2048, C=1024, H=16, Dh=64.
// Pipeline: cvt/transposes -> QKV GEMM (double-buffered async-LDS staging,
//           WMMA) -> V transpose -> flash attention (WMMA) -> out GEMM (WMMA).
// ---------------------------------------------------------------------------

typedef __attribute__((ext_vector_type(16))) __bf16 v16bf;
typedef __attribute__((ext_vector_type(8)))  float  v8f;

union FragBF16 {
  v16bf v;
  uint4 q[2];
  __bf16 e[16];
};

static __device__ inline v8f vzero8() {
  v8f z;
#pragma unroll
  for (int i = 0; i < 8; ++i) z[i] = 0.0f;
  return z;
}

static __device__ inline v8f wmma_bf16(const FragBF16& a, const FragBF16& b, v8f c) {
  // v_wmma_f32_16x16x32_bf16  D = A(16x32) * B(32x16) + C(16x16 f32)
  return __builtin_amdgcn_wmma_f32_16x16x32_bf16(false, a.v, false, b.v,
                                                 (short)0, c, false, false);
}

// --- CDNA5 async copy: 16 bytes global -> LDS, tracked on ASYNCcnt ---------
// Generic pointers to __shared__ carry the LDS byte offset in the low 32 bits
static __device__ inline void async_ld_b128(void* lds, const void* g) {
  asm volatile("global_load_async_to_lds_b128 %0, %1, off"
               :: "v"((unsigned)(uintptr_t)lds), "v"(g)
               : "memory");
}
// Async loads complete in order; each GEMM tile is 4 async instructions per
// wave, so waiting ASYNCcnt<=4 means "previous tile has fully landed".
static __device__ inline void wait_asynccnt0() {
  asm volatile("s_wait_asynccnt 0x0" ::: "memory");
}
static __device__ inline void wait_asynccnt4() {
  asm volatile("s_wait_asynccnt 0x4" ::: "memory");
}

// ---------------------------------------------------------------------------
// f32 -> bf16 conversion (straight copy)
// ---------------------------------------------------------------------------
__global__ void cvt_f32_to_bf16_kernel(const float* __restrict__ in,
                                       unsigned short* __restrict__ out, int n) {
  int i = blockIdx.x * blockDim.x + threadIdx.x;
  if (i < n) {
    reinterpret_cast<__bf16*>(out)[i] = (__bf16)in[i];
  }
}

// ---------------------------------------------------------------------------
// f32 [K,N] -> bf16 [N,K] transpose (32x32 LDS tile, both sides coalesced)
// ---------------------------------------------------------------------------
__global__ __launch_bounds__(256) void cvt_transpose_kernel(
    const float* __restrict__ in, unsigned short* __restrict__ outraw,
    int K, int N) {
  __bf16* out = reinterpret_cast<__bf16*>(outraw);
  __shared__ __bf16 tile[32][33];
  const int tx = threadIdx.x & 31;
  const int ty = threadIdx.x >> 5;  // 0..7
  const int n0 = blockIdx.x * 32;
  const int k0 = blockIdx.y * 32;
#pragma unroll
  for (int i = 0; i < 4; ++i) {
    const int kl = ty + i * 8;
    tile[kl][tx] = (__bf16)in[(size_t)(k0 + kl) * N + n0 + tx];
  }
  __syncthreads();
#pragma unroll
  for (int i = 0; i < 4; ++i) {
    const int nl = ty + i * 8;
    out[(size_t)(n0 + nl) * K + k0 + tx] = tile[tx][nl];
  }
}

// ---------------------------------------------------------------------------
// bf16 V slice of qkv [B*T, 3072] -> vt [B*H*64, T]  (32x32 LDS tile)
// ---------------------------------------------------------------------------
__global__ __launch_bounds__(256) void transpose_v_kernel(
    const unsigned short* __restrict__ qkvraw,
    unsigned short* __restrict__ vtraw) {
  constexpr int T = 2048, R3C = 3072;
  const __bf16* qkv = reinterpret_cast<const __bf16*>(qkvraw);
  __bf16* vt = reinterpret_cast<__bf16*>(vtraw);
  __shared__ __bf16 tile[32][33];
  const int tx = threadIdx.x & 31;
  const int ty = threadIdx.x >> 5;
  const int t0 = blockIdx.x * 32;    // 64 tiles along T
  const int bhd = blockIdx.y;        // b*32 + h*2 + dtile  (128)
  const int b  = bhd >> 5;
  const int h  = (bhd >> 1) & 15;
  const int d0 = (bhd & 1) * 32;
  const size_t srcBase = (size_t)b * T * R3C + 2048 + h * 64 + d0;
#pragma unroll
  for (int i = 0; i < 4; ++i) {
    const int tl = ty + i * 8;
    tile[tl][tx] = qkv[srcBase + (size_t)(t0 + tl) * R3C + tx];  // [t][d]
  }
  __syncthreads();
  const size_t dstBase = (size_t)((b * 16 + h) * 64 + d0) * T;
#pragma unroll
  for (int i = 0; i < 4; ++i) {
    const int dl = ty + i * 8;
    vt[dstBase + (size_t)dl * T + t0 + tx] = tile[tx][dl];
  }
}

// ---------------------------------------------------------------------------
// Tiled bf16 GEMM: C[M,N] = A[M,K] * Bt[N,K]^T + bias[N]
// Block tile 128x128, 256 threads (8 waves), wave tile 32x64, K-step 32.
// Double-buffered: tile i+1 streams into LDS via global_load_async_to_lds_b128
// (ASYNCcnt) while tile i feeds the WMMA pipe. All fragments are contiguous
// b128 LDS loads with k = 16*(lane/16) + e, the same order for A and B.
// ---------------------------------------------------------------------------
template <bool OUT_BF16>
__global__ __launch_bounds__(256) void gemm_bf16_kernel(
    const unsigned short* __restrict__ Araw,
    const unsigned short* __restrict__ Btraw,
    const float* __restrict__ bias,
    void* __restrict__ Cout,
    int M, int N, int K) {
  const __bf16* A  = reinterpret_cast<const __bf16*>(Araw);
  const __bf16* Bt = reinterpret_cast<const __bf16*>(Btraw);

  __shared__ __align__(16) __bf16 As[2][128 * 32];  // [buf][row][k] 2x8 KB
  __shared__ __align__(16) __bf16 Bs[2][128 * 32];  // [buf][col][k] 2x8 KB

  const int tid   = threadIdx.x;
  const int lane  = tid & 31;
  const int wave  = tid >> 5;
  const int waveM = wave & 3;
  const int waveN = wave >> 2;
  const int n16   = lane & 15;
  const int hl    = lane >> 4;

  const int blockN = blockIdx.x * 128;
  const int blockM = blockIdx.y * 128;

  v8f acc[2][4];
#pragma unroll
  for (int mi = 0; mi < 2; ++mi)
#pragma unroll
    for (int ni = 0; ni < 4; ++ni) acc[mi][ni] = vzero8();

  const int rowS = tid >> 1;        // 0..127
  const int ccS  = (tid & 1) * 16;  // 0 or 16

  // issue one tile's async stage (4 b128 per lane => ASYNCcnt += 4 per wave)
  auto issue_tile = [&](int buf, int k0) {
    const __bf16* srcA = A + (size_t)(blockM + rowS) * K + k0 + ccS;
    async_ld_b128(&As[buf][rowS * 32 + ccS],     srcA);
    async_ld_b128(&As[buf][rowS * 32 + ccS + 8], srcA + 8);
    const __bf16* srcB = Bt + (size_t)(blockN + rowS) * K + k0 + ccS;
    async_ld_b128(&Bs[buf][rowS * 32 + ccS],     srcB);
    async_ld_b128(&Bs[buf][rowS * 32 + ccS + 8], srcB + 8);
  };

  const int nTiles = K >> 5;
  issue_tile(0, 0);

  for (int i = 0; i < nTiles; ++i) {
    const int cur = i & 1;
    if (i + 1 < nTiles) {
      issue_tile(cur ^ 1, (i + 1) << 5);  // prefetch next tile
      wait_asynccnt4();                   // tile i has landed (in-order)
    } else {
      wait_asynccnt0();
    }
    __syncthreads();

    // --- fragments from LDS (all contiguous b128) ---
    FragBF16 a[2];
#pragma unroll
    for (int mi = 0; mi < 2; ++mi) {
      const int row = waveM * 32 + mi * 16 + n16;
      const uint4* p = reinterpret_cast<const uint4*>(&As[cur][row * 32]);
      a[mi].q[0] = p[2 * hl];      // e -> K = hl*16 + e
      a[mi].q[1] = p[2 * hl + 1];
    }
    FragBF16 b[4];
#pragma unroll
    for (int ni = 0; ni < 4; ++ni) {
      const int col = waveN * 64 + ni * 16 + n16;
      const uint4* p = reinterpret_cast<const uint4*>(&Bs[cur][col * 32]);
      b[ni].q[0] = p[2 * hl];      // e -> K = hl*16 + e
      b[ni].q[1] = p[2 * hl + 1];
    }
#pragma unroll
    for (int mi = 0; mi < 2; ++mi)
#pragma unroll
      for (int ni = 0; ni < 4; ++ni)
        acc[mi][ni] = wmma_bf16(a[mi], b[ni], acc[mi][ni]);

    __syncthreads();  // all waves done reading buf[cur] before its reuse
  }

  // --- epilogue: bias + store (C layout: row = r + 8*hl, col = n16) ---
#pragma unroll
  for (int mi = 0; mi < 2; ++mi)
#pragma unroll
    for (int ni = 0; ni < 4; ++ni) {
      const int col = blockN + waveN * 64 + ni * 16 + n16;
      const float bv = bias ? bias[col] : 0.0f;
#pragma unroll
      for (int r = 0; r < 8; ++r) {
        const int row = blockM + waveM * 32 + mi * 16 + r + 8 * hl;
        const float vv = acc[mi][ni][r] + bv;
        if (OUT_BF16)
          reinterpret_cast<__bf16*>(Cout)[(size_t)row * N + col] = (__bf16)vv;
        else
          reinterpret_cast<float*>(Cout)[(size_t)row * N + col] = vv;
      }
    }
}

// ---------------------------------------------------------------------------
// Flash attention (causal, online softmax). One wave per 16-query tile.
// qkv: bf16 [B*T, 3*C] (token rows -> Q/K rows are 64-elem d-contiguous)
// vt:  bf16 [B*H*64, T] (pre-transposed V -> B-fragments key-contiguous)
// y:   bf16 [B*T, C]
// ---------------------------------------------------------------------------
__global__ __launch_bounds__(32) void flash_attn_kernel(
    const unsigned short* __restrict__ qkvraw,
    const unsigned short* __restrict__ vtraw,
    unsigned short* __restrict__ yraw) {
  constexpr int T = 2048, C = 1024, R3C = 3072;
  const __bf16* qkv = reinterpret_cast<const __bf16*>(qkvraw);
  const __bf16* vt  = reinterpret_cast<const __bf16*>(vtraw);
  __bf16* y = reinterpret_cast<__bf16*>(yraw);

  const int qt = blockIdx.x & (T / 16 - 1);
  const int bh = blockIdx.x >> 7;
  const int b  = bh >> 4;
  const int h  = bh & 15;
  const int lane = threadIdx.x;
  const int n16  = lane & 15;
  const int hl   = lane >> 4;
  const int t0   = qt * 16;

  const size_t tokBase = (size_t)b * T;
  const __bf16* qp  = qkv + tokBase * R3C + 0 * C + h * 64;
  const __bf16* kp  = qkv + tokBase * R3C + 1 * C + h * 64;
  const __bf16* vtp = vt + (size_t)bh * 64 * T;  // rows = d, stride T

  __shared__ __align__(16) __bf16 Plds[16 * 32];  // P tile, [row][key]

  // Q fragments for d-chunks {0..31},{32..63}
  FragBF16 qa[2];
#pragma unroll
  for (int dc = 0; dc < 2; ++dc) {
    const uint4* p =
        reinterpret_cast<const uint4*>(qp + (size_t)(t0 + n16) * R3C + dc * 32);
    qa[dc].q[0] = p[2 * hl];
    qa[dc].q[1] = p[2 * hl + 1];
  }

  v8f accO[4];
#pragma unroll
  for (int ni = 0; ni < 4; ++ni) accO[ni] = vzero8();
  float m_i[8], l_i[8];
#pragma unroll
  for (int r = 0; r < 8; ++r) { m_i[r] = -INFINITY; l_i[r] = 0.0f; }

  const float scale = 0.125f;  // 1/sqrt(64)

  for (int s = 0; s < t0 + 16; s += 32) {
    // K fragments (B layout): lane%16 = key, k-dim = d
    FragBF16 kb[2][2];
#pragma unroll
    for (int dc = 0; dc < 2; ++dc)
#pragma unroll
      for (int j = 0; j < 2; ++j) {
        const uint4* p = reinterpret_cast<const uint4*>(
            kp + (size_t)(s + j * 16 + n16) * R3C + dc * 32);
        kb[dc][j].q[0] = p[2 * hl];
        kb[dc][j].q[1] = p[2 * hl + 1];
      }

    v8f S0 = wmma_bf16(qa[0], kb[0][0], wmma_bf16(qa[1], kb[1][0], vzero8()));
    v8f S1 = wmma_bf16(qa[0], kb[0][1], wmma_bf16(qa[1], kb[1][1], vzero8()));

    // scale + causal mask + online softmax (row = t0 + r + 8*hl)
    float p0[8], p1[8], alpha[8];
#pragma unroll
    for (int r = 0; r < 8; ++r) {
      const int rowg = t0 + r + 8 * hl;
      float s0 = S0[r] * scale;
      float s1 = S1[r] * scale;
      if (s + n16 > rowg) s0 = -INFINITY;
      if (s + 16 + n16 > rowg) s1 = -INFINITY;
      float mx = fmaxf(s0, s1);
#pragma unroll
      for (int off = 8; off > 0; off >>= 1)
        mx = fmaxf(mx, __shfl_xor(mx, off, 16));
      const float nm = fmaxf(m_i[r], mx);
      const float al = __expf(m_i[r] - nm);
      p0[r] = __expf(s0 - nm);
      p1[r] = __expf(s1 - nm);
      float rs = p0[r] + p1[r];
#pragma unroll
      for (int off = 8; off > 0; off >>= 1) rs += __shfl_xor(rs, off, 16);
      l_i[r] = l_i[r] * al + rs;
      m_i[r] = nm;
      alpha[r] = al;
    }
#pragma unroll
    for (int ni = 0; ni < 4; ++ni)
#pragma unroll
      for (int r = 0; r < 8; ++r) accO[ni][r] *= alpha[r];

    // P (C layout) -> LDS, reload as A fragment
#pragma unroll
    for (int r = 0; r < 8; ++r) {
      const int row = r + 8 * hl;
      Plds[row * 32 + n16]      = (__bf16)p0[r];
      Plds[row * 32 + 16 + n16] = (__bf16)p1[r];
    }
    __syncthreads();

    FragBF16 pa;
    {
      const uint4* p = reinterpret_cast<const uint4*>(Plds + n16 * 32);
      pa.q[0] = p[2 * hl];
      pa.q[1] = p[2 * hl + 1];
    }
    // V B-fragments straight from pre-transposed global Vt (key-contiguous)
    FragBF16 vb[4];
#pragma unroll
    for (int ni = 0; ni < 4; ++ni) {
      const uint4* p = reinterpret_cast<const uint4*>(
          vtp + (size_t)(ni * 16 + n16) * T + s + hl * 16);
      vb[ni].q[0] = p[0];
      vb[ni].q[1] = p[1];
    }
#pragma unroll
    for (int ni = 0; ni < 4; ++ni) accO[ni] = wmma_bf16(pa, vb[ni], accO[ni]);
    __syncthreads();
  }

  // normalize and write y (bf16)
#pragma unroll
  for (int ni = 0; ni < 4; ++ni)
#pragma unroll
    for (int r = 0; r < 8; ++r) {
      const int rowg = t0 + r + 8 * hl;
      const float ov = accO[ni][r] / l_i[r];
      y[(tokBase + rowg) * C + h * 64 + ni * 16 + n16] = (__bf16)ov;
    }
}

// ---------------------------------------------------------------------------
// Launch
// ---------------------------------------------------------------------------
extern "C" void kernel_launch(void* const* d_in, const int* in_sizes, int n_in,
                              void* d_out, int out_size, void* d_ws, size_t ws_size,
                              hipStream_t stream) {
  constexpr int B = 4, T = 2048, C = 1024, H = 16;
  constexpr int M = B * T;   // 8192
  constexpr int N1 = 3 * C;  // 3072
  constexpr size_t XB_OFF     = 0;
  constexpr size_t WQKVT_OFF  = XB_OFF + (size_t)M * C * 2;
  constexpr size_t WPROJT_OFF = WQKVT_OFF + (size_t)C * N1 * 2;
  constexpr size_t QKV_OFF    = WPROJT_OFF + (size_t)C * C * 2;
  constexpr size_t VT_OFF     = QKV_OFF + (size_t)M * N1 * 2;
  constexpr size_t YB_OFF     = VT_OFF + (size_t)B * H * 64 * T * 2;
  constexpr size_t NEEDED     = YB_OFF + (size_t)M * C * 2;  // ~109 MB
  if (ws_size < NEEDED) return;

  const float* x     = (const float*)d_in[0];
  const float* Wqkv  = (const float*)d_in[1];
  const float* bqkv  = (const float*)d_in[2];
  const float* Wproj = (const float*)d_in[3];
  const float* bproj = (const float*)d_in[4];

  char* ws = (char*)d_ws;
  unsigned short* xb     = (unsigned short*)(ws + XB_OFF);
  unsigned short* wqkvt  = (unsigned short*)(ws + WQKVT_OFF);
  unsigned short* wprojt = (unsigned short*)(ws + WPROJT_OFF);
  unsigned short* qkvb   = (unsigned short*)(ws + QKV_OFF);
  unsigned short* vtb    = (unsigned short*)(ws + VT_OFF);
  unsigned short* yb     = (unsigned short*)(ws + YB_OFF);

  // 1) conversions: x straight; weights converted AND transposed to [N,K]
  {
    const int n = M * C;
    cvt_f32_to_bf16_kernel<<<(n + 255) / 256, 256, 0, stream>>>(x, xb, n);
  }
  cvt_transpose_kernel<<<dim3(N1 / 32, C / 32), 256, 0, stream>>>(Wqkv, wqkvt, C, N1);
  cvt_transpose_kernel<<<dim3(C / 32, C / 32), 256, 0, stream>>>(Wproj, wprojt, C, C);

  // 2) qkv = x @ Wqkv + bqkv   (bf16 out)
  gemm_bf16_kernel<true><<<dim3(N1 / 128, M / 128), 256, 0, stream>>>(
      xb, wqkvt, bqkv, qkvb, M, N1, C);

  // 3) V transpose for key-contiguous P*V fragments
  transpose_v_kernel<<<dim3(T / 32, B * H * 2), 256, 0, stream>>>(qkvb, vtb);

  // 4) flash attention -> y (bf16)
  flash_attn_kernel<<<dim3(B * H * (T / 16)), 32, 0, stream>>>(qkvb, vtb, yb);

  // 5) out = y @ Wproj + bproj (f32 out)
  gemm_bf16_kernel<false><<<dim3(C / 128, M / 128), 256, 0, stream>>>(
      yb, wprojt, bproj, (float*)d_out, M, C, C);
}
